// TDMHSA_74715251081240
// MI455X (gfx1250) — compile-verified, matrix operand order checked by reading, not run on hardware
//
#include <hip/hip_runtime.h>
#include <hip/hip_bf16.h>

typedef _Float16 half_t;
typedef __attribute__((ext_vector_type(16))) _Float16 v16h;
typedef __attribute__((ext_vector_type(8)))  _Float16 v8h;
typedef __attribute__((ext_vector_type(8)))  float    v8f;

// Sizes (fixed by the reference)
#define B_  8
#define N_  1024
#define C_  1024
#define H_  16
#define D_  64
#define T_  256

// ---------------------------------------------------------------------------
// TDM support (guarded: 5-arg on ROCm7.2/clang-22, 6-arg on clang-23,
// cooperative-load fallback otherwise).
// ---------------------------------------------------------------------------
#if defined(__HIP_DEVICE_COMPILE__)
#if __has_builtin(__builtin_amdgcn_tensor_load_to_lds) && \
    __has_builtin(__builtin_amdgcn_s_wait_tensorcnt)
#define USE_TDM 1
#endif
#endif

#ifdef USE_TDM
typedef __attribute__((ext_vector_type(4))) unsigned int u32x4_t;
typedef __attribute__((ext_vector_type(4))) int i32x4_t;
typedef __attribute__((ext_vector_type(8))) int i32x8_t;

// Issue a TDM load of a 64-row x 32-col f16 tile (row stride = stride_elems)
// from global into LDS at byte offset lds_off. One call per wave; EXEC ignored.
// D# packing per CDNA5 ISA 8.3/8.4: group0 = {count=1, lds_addr,
// global_addr[56:0], type=2}; group1 = {mask=0, data_size=1(2B),
// tensor_dim0=32, tensor_dim1=64, tile_dim0=32, tile_dim1=64, stride}.
__device__ __forceinline__ void tdm_load_tile_64x32(unsigned int lds_off,
                                                    const half_t* gsrc,
                                                    unsigned int stride_elems) {
  unsigned long long ga = (unsigned long long)gsrc;
  unsigned int ga_lo = (unsigned int)__builtin_amdgcn_readfirstlane((int)(unsigned int)ga);
  unsigned int ga_hi = (unsigned int)__builtin_amdgcn_readfirstlane((int)(unsigned int)(ga >> 32));
  unsigned int loff  = (unsigned int)__builtin_amdgcn_readfirstlane((int)lds_off);
  unsigned int strd  = (unsigned int)__builtin_amdgcn_readfirstlane((int)stride_elems);
  u32x4_t g0;
  g0[0] = 1u;                                   // count=1 (valid user D#)
  g0[1] = loff;                                 // lds_addr (bytes)
  g0[2] = ga_lo;                                // global_addr[31:0]
  g0[3] = (ga_hi & 0x01FFFFFFu) | (2u << 30);   // global_addr[56:32] | type=2
  i32x8_t g1;
  g1[0] = 0x00010000;        // workgroup_mask=0 | data_size=1 (2 bytes)
  g1[1] = (int)(32u << 16);  // tensor_dim0[15:0] = 32
  g1[2] = (int)(64u << 16);  // tensor_dim0[31:16]=0 | tensor_dim1[15:0] = 64
  g1[3] = (int)(32u << 16);  // tensor_dim1[31:16]=0 | tile_dim0 = 32
  g1[4] = 64;                // tile_dim1 = 64 | tile_dim2 = 0
  g1[5] = (int)strd;         // tensor_dim0_stride[31:0]
  g1[6] = 0;                 // stride[47:32] | tensor_dim1_stride[15:0] = 0
  g1[7] = 0;
  i32x4_t z4 = {0, 0, 0, 0};
#if __clang_major__ >= 23
  i32x8_t z8 = {0, 0, 0, 0, 0, 0, 0, 0};
  __builtin_amdgcn_tensor_load_to_lds(g0, g1, z4, z4, z8, 0);
#else
  __builtin_amdgcn_tensor_load_to_lds(g0, g1, z4, z4, 0);
#endif
}
#endif  // USE_TDM

// ---------------------------------------------------------------------------
// Kernel 0: fp32 -> f16 conversion (x, W_qkv, W_out, Er), one pass over HBM.
// ---------------------------------------------------------------------------
__global__ __launch_bounds__(256) void cvt_kernel(
    const float* __restrict__ s, half_t* __restrict__ d, int n) {
  int i = blockIdx.x * 256 + threadIdx.x;
  if (i < n) d[i] = (half_t)s[i];
}

// ---------------------------------------------------------------------------
// Kernel 1: tqkv[b, j] = time_emb[b] @ W_time[:, j] + b_time[j]   (8 x 3072)
// ---------------------------------------------------------------------------
__global__ __launch_bounds__(256) void time_proj_kernel(
    const float* __restrict__ te, const float* __restrict__ Wt,
    const float* __restrict__ bt, float* __restrict__ tqkv) {
  int idx = blockIdx.x * 256 + threadIdx.x;
  if (idx >= B_ * 3 * C_) return;
  int b = idx / (3 * C_), j = idx % (3 * C_);
  float acc = bt[j];
  for (int t = 0; t < T_; ++t) acc += te[b * T_ + t] * Wt[t * (3 * C_) + j];
  tqkv[idx] = acc;
}

// ---------------------------------------------------------------------------
// Kernel 2: qkv = x @ W_qkv + b_qkv + tqkv[b]  (8192 x 3072, K=1024), f16 WMMA.
// A tile via TDM (64x32 f16), B tile cooperative transposed store.
// Scatters q,k row-major f16 and v transposed f16.
// ---------------------------------------------------------------------------
__global__ __launch_bounds__(128) void qkv_gemm_kernel(
    const half_t* __restrict__ xh, const half_t* __restrict__ Wh,
    const float* __restrict__ bias, const float* __restrict__ tqkv,
    half_t* __restrict__ qh, half_t* __restrict__ kh, half_t* __restrict__ vTh) {
  __shared__ half_t As[64][32];    // A tile, row-major
  __shared__ half_t BsT[64][32];   // B tile stored transposed: BsT[n][k]
  const int r0 = blockIdx.x * 64, c0 = blockIdx.y * 64;
  const int tid = threadIdx.x, wave = tid >> 5, lane = tid & 31;
  const int M = lane & 15, hi = lane >> 4;
  v8f acc[4] = {};

  for (int kk = 0; kk < C_; kk += 32) {
    __syncthreads();
#ifdef USE_TDM
    if (wave == 0)
      tdm_load_tile_64x32((unsigned int)(unsigned long long)&As[0][0],
                          xh + (size_t)r0 * C_ + kk, C_);
#else
#pragma unroll
    for (int t = 0; t < 16; ++t) {
      int idx = tid * 16 + t;
      int ar = idx >> 5, ac = idx & 31;
      As[ar][ac] = xh[(size_t)(r0 + ar) * C_ + kk + ac];
    }
#endif
#pragma unroll
    for (int t = 0; t < 16; ++t) {
      int idx = tid * 16 + t;                   // 0..2047
      int bk = idx >> 6, bn = idx & 63;
      BsT[bn][bk] = Wh[(size_t)(kk + bk) * (3 * C_) + c0 + bn];
    }
    if (kk + 32 < C_) {   // cover next tile's latency (global_prefetch_b8)
      int idx = tid * 16;
      __builtin_prefetch(Wh + (size_t)(kk + 32 + (idx >> 6)) * (3 * C_) + c0 + (idx & 63), 0, 1);
      __builtin_prefetch(xh + (size_t)(r0 + (idx >> 5)) * C_ + kk + 32, 0, 1);
    }
#ifdef USE_TDM
    if (wave == 0) __builtin_amdgcn_s_wait_tensorcnt(0);
#endif
    __syncthreads();
    // A fragment: a[e] = A[M][8*hi+e], a[8+e] = A[M][16+8*hi+e]   (ISA 7.12.2)
    v16h a;
    ((v8h*)&a)[0] = *(const v8h*)&As[wave * 16 + M][hi * 8];
    ((v8h*)&a)[1] = *(const v8h*)&As[wave * 16 + M][16 + hi * 8];
#pragma unroll
    for (int t = 0; t < 4; ++t) {
      v16h bf;
      ((v8h*)&bf)[0] = *(const v8h*)&BsT[t * 16 + M][hi * 8];
      ((v8h*)&bf)[1] = *(const v8h*)&BsT[t * 16 + M][16 + hi * 8];
      acc[t] = __builtin_amdgcn_wmma_f32_16x16x32_f16(false, a, false, bf,
                                                      (short)0, acc[t], false, false);
    }
  }
  // Epilogue: add biases, scatter into q/k (B,H,N,D) and vT (B,H,D,N) as f16
#pragma unroll
  for (int t = 0; t < 4; ++t) {
#pragma unroll
    for (int v = 0; v < 8; ++v) {
      int g = r0 + wave * 16 + v + 8 * hi;     // global row = b*N + n
      int j = c0 + t * 16 + M;                 // global col in [0,3072)
      float val = acc[t][v] + bias[j] + tqkv[(g >> 10) * (3 * C_) + j];
      int b = g >> 10, n = g & (N_ - 1);
      int sel = j >> 10, h = (j >> 6) & 15, d = j & 63;
      half_t hv = (half_t)val;
      size_t bh = (size_t)(b * H_ + h);
      if (sel == 0)      qh[(bh * N_ + n) * D_ + d] = hv;
      else if (sel == 1) kh[(bh * N_ + n) * D_ + d] = hv;
      else               vTh[(bh * D_ + d) * N_ + n] = hv;
    }
  }
}

// ---------------------------------------------------------------------------
// Kernel 3: fused relative attention. One wave (32 thr) per (b, h, 16 rows).
// LDS: 16x1024 f16 score strip (reused in place for softmax probs).
// Skew inverse map: rel[i',m] -> (i', m-1023+i') if m>=1023-i' else (i'-1, m+i'+1)
// ---------------------------------------------------------------------------
__global__ __launch_bounds__(32) void attn_kernel(
    const half_t* __restrict__ qh, const half_t* __restrict__ kh,
    const half_t* __restrict__ vTh, const half_t* __restrict__ Erh,
    half_t* __restrict__ oh) {
  __shared__ half_t sc[16][N_];   // 32 KB
  __shared__ half_t qx[D_];       // q row i0+16 for boundary rel
  const int i0 = blockIdx.x * 16, h = blockIdx.y, b = blockIdx.z;
  const int lane = threadIdx.x, M = lane & 15, hi = lane >> 4;
  const size_t bh = (size_t)(b * H_ + h);
  const half_t* qbase = qh + bh * N_ * D_;
  const half_t* kbase = kh + bh * N_ * D_;
  const half_t* vbase = vTh + bh * D_ * N_;

  // Persistent A fragments for q rows i0..i0+15 (two K-slabs of 32 over D=64)
  v16h qa[2];
#pragma unroll
  for (int kf = 0; kf < 2; ++kf) {
    const half_t* p = qbase + (size_t)(i0 + M) * D_ + kf * 32 + hi * 8;
    ((v8h*)&qa[kf])[0] = *(const v8h*)p;
    ((v8h*)&qa[kf])[1] = *(const v8h*)(p + 16);
  }
  if (i0 + 16 < N_) {
    qx[lane * 2]     = qbase[(size_t)(i0 + 16) * D_ + lane * 2];
    qx[lane * 2 + 1] = qbase[(size_t)(i0 + 16) * D_ + lane * 2 + 1];
  } else { qx[lane * 2] = (half_t)0.f; qx[lane * 2 + 1] = (half_t)0.f; }

  // Phase 1: S = (q k^T) * D^-1/2  into LDS
#pragma unroll 1
  for (int c0 = 0; c0 < N_; c0 += 16) {
    if (c0 + 16 < N_)
      __builtin_prefetch(kbase + (size_t)(c0 + 16 + M) * D_ + hi * 8, 0, 1);
    v8f acc = {};
#pragma unroll
    for (int kf = 0; kf < 2; ++kf) {
      v16h bf;
      const half_t* p = kbase + (size_t)(c0 + M) * D_ + kf * 32 + hi * 8;
      ((v8h*)&bf)[0] = *(const v8h*)p;
      ((v8h*)&bf)[1] = *(const v8h*)(p + 16);
      acc = __builtin_amdgcn_wmma_f32_16x16x32_f16(false, qa[kf], false, bf,
                                                   (short)0, acc, false, false);
    }
#pragma unroll
    for (int v = 0; v < 8; ++v)
      sc[v + 8 * hi][c0 + M] = (half_t)(acc[v] * 0.125f);
  }
  __syncthreads();

  // Phase 2: rel = q @ Er^T tiles, scatter-added via the skew inverse map
#pragma unroll 1
  for (int m0 = 0; m0 < N_; m0 += 16) {
    v8f acc = {};
#pragma unroll
    for (int kf = 0; kf < 2; ++kf) {
      v16h bf;
      const half_t* p = Erh + (size_t)(m0 + M) * D_ + kf * 32 + hi * 8;
      ((v8h*)&bf)[0] = *(const v8h*)p;
      ((v8h*)&bf)[1] = *(const v8h*)(p + 16);
      acc = __builtin_amdgcn_wmma_f32_16x16x32_f16(false, qa[kf], false, bf,
                                                   (short)0, acc, false, false);
    }
    int m = m0 + M;
#pragma unroll
    for (int v = 0; v < 8; ++v) {
      int r = v + 8 * hi, iq = i0 + r;
      if (m >= (N_ - 1) - iq) {              // lower triangle: row iq, col m-1023+iq
        int c = m - (N_ - 1) + iq;
        sc[r][c] = (half_t)((float)sc[r][c] + acc[v]);
      } else if (r >= 1) {                   // upper: row iq-1, col m+iq+1
        int c = m + iq + 1;
        sc[r - 1][c] = (half_t)((float)sc[r - 1][c] + acc[v]);
      }
    }
  }
  // Phase 2b: boundary source row i' = i0+16 contributes to score row 15
  if (i0 + 16 < N_) {
    float qf[D_];
#pragma unroll
    for (int d = 0; d < D_; ++d) qf[d] = (float)qx[d];
#pragma unroll 1
    for (int t = 0; t < 32; ++t) {
      int m = lane + 32 * t;
      int c = m + i0 + 17;
      if (c < N_) {
        const half_t* er = Erh + (size_t)m * D_;
        float dot = 0.f;
#pragma unroll
        for (int d = 0; d < D_; ++d) dot += qf[d] * (float)er[d];
        sc[15][c] = (half_t)((float)sc[15][c] + dot);
      }
    }
  }
  __syncthreads();

  // Phase 3: softmax. 2 lanes per row, each owns a disjoint 512-col half;
  // probabilities overwrite scores element-for-element (same address).
  float rowsum;
  {
    const int r = M, cb = hi * 512;
    float mx = -1e30f;
    for (int c = 0; c < 512; ++c) mx = fmaxf(mx, (float)sc[r][cb + c]);
    mx = fmaxf(mx, __shfl(mx, lane ^ 16, 32));
    float sum = 0.f;
    for (int c = 0; c < 512; ++c) {
      float p = __expf((float)sc[r][cb + c] - mx);
      sum += p;
      sc[r][cb + c] = (half_t)p;
    }
    rowsum = sum + __shfl(sum, lane ^ 16, 32);
  }
  __syncthreads();

  // Phase 4: O = P @ v  (A from LDS f16 probs, B from transposed v)
  v8f oacc[4] = {};
#pragma unroll 1
  for (int k0 = 0; k0 < N_; k0 += 32) {
    v16h a;
    const half_t* p = &sc[M][k0 + hi * 8];
    ((v8h*)&a)[0] = *(const v8h*)p;
    ((v8h*)&a)[1] = *(const v8h*)(p + 16);
#pragma unroll
    for (int t = 0; t < 4; ++t) {
      v16h bf;
      const half_t* vp = vbase + (size_t)(t * 16 + M) * N_ + k0 + hi * 8;
      ((v8h*)&bf)[0] = *(const v8h*)vp;
      ((v8h*)&bf)[1] = *(const v8h*)(vp + 16);
      oacc[t] = __builtin_amdgcn_wmma_f32_16x16x32_f16(false, a, false, bf,
                                                       (short)0, oacc[t], false, false);
    }
  }
  // Normalize by row sums and store to O in (B, N, H*D) layout (f16)
#pragma unroll
  for (int v = 0; v < 8; ++v) {
    float inv = 1.f / __shfl(rowsum, v + 8 * hi, 32);
    int i = i0 + v + 8 * hi;
#pragma unroll
    for (int t = 0; t < 4; ++t) {
      int d = t * 16 + M;
      oh[(((size_t)b * N_ + i) * H_ + h) * D_ + d] = (half_t)(oacc[t][v] * inv);
    }
  }
}

// ---------------------------------------------------------------------------
// Kernel 4: out = O @ W_out + b_out  (8192 x 1024, K=1024), fp32 result.
// A tile via TDM from the f16 attention output.
// ---------------------------------------------------------------------------
__global__ __launch_bounds__(128) void out_gemm_kernel(
    const half_t* __restrict__ A, const half_t* __restrict__ Wh,
    const float* __restrict__ bias, float* __restrict__ out) {
  __shared__ half_t As[64][32];
  __shared__ half_t BsT[64][32];
  const int r0 = blockIdx.x * 64, c0 = blockIdx.y * 64;
  const int tid = threadIdx.x, wave = tid >> 5, lane = tid & 31;
  const int M = lane & 15, hi = lane >> 4;
  v8f acc[4] = {};

  for (int kk = 0; kk < C_; kk += 32) {
    __syncthreads();
#ifdef USE_TDM
    if (wave == 0)
      tdm_load_tile_64x32((unsigned int)(unsigned long long)&As[0][0],
                          A + (size_t)r0 * C_ + kk, C_);
#else
#pragma unroll
    for (int t = 0; t < 16; ++t) {
      int idx = tid * 16 + t;
      int ar = idx >> 5, ac = idx & 31;
      As[ar][ac] = A[(size_t)(r0 + ar) * C_ + kk + ac];
    }
#endif
#pragma unroll
    for (int t = 0; t < 16; ++t) {
      int idx = tid * 16 + t;
      int bk = idx >> 6, bn = idx & 63;
      BsT[bn][bk] = Wh[(size_t)(kk + bk) * C_ + c0 + bn];
    }
    if (kk + 32 < C_) {
      int idx = tid * 16;
      __builtin_prefetch(Wh + (size_t)(kk + 32 + (idx >> 6)) * C_ + c0 + (idx & 63), 0, 1);
      __builtin_prefetch(A + (size_t)(r0 + (idx >> 5)) * C_ + kk + 32, 0, 1);
    }
#ifdef USE_TDM
    if (wave == 0) __builtin_amdgcn_s_wait_tensorcnt(0);
#endif
    __syncthreads();
    v16h a;
    ((v8h*)&a)[0] = *(const v8h*)&As[wave * 16 + M][hi * 8];
    ((v8h*)&a)[1] = *(const v8h*)&As[wave * 16 + M][16 + hi * 8];
#pragma unroll
    for (int t = 0; t < 4; ++t) {
      v16h bf;
      ((v8h*)&bf)[0] = *(const v8h*)&BsT[t * 16 + M][hi * 8];
      ((v8h*)&bf)[1] = *(const v8h*)&BsT[t * 16 + M][16 + hi * 8];
      acc[t] = __builtin_amdgcn_wmma_f32_16x16x32_f16(false, a, false, bf,
                                                      (short)0, acc[t], false, false);
    }
  }
#pragma unroll
  for (int t = 0; t < 4; ++t) {
#pragma unroll
    for (int v = 0; v < 8; ++v) {
      int g = r0 + wave * 16 + v + 8 * hi;
      int j = c0 + t * 16 + M;
      out[(size_t)g * C_ + j] = acc[t][v] + bias[j];
    }
  }
}

// ---------------------------------------------------------------------------
// Launcher. Inputs: x, time_emb, W_qkv, b_qkv, W_time, b_time, W_out, b_out, Er
// Workspace (bytes):
//   [0, 96K)       tqkv fp32 (8 x 3072)
//   [128K, +16M)   q   f16 (B,H,N,D)
//   [+16M]         k   f16 (B,H,N,D)
//   [+16M]         vT  f16 (B,H,D,N)
//   [+16M]         O   f16 (B,N,H,D)
//   [+16M]         x   f16
//   [+6M]          W_qkv f16
//   [+2M]          W_out f16
//   [+128K]        Er  f16            total ~88.3 MB
// ---------------------------------------------------------------------------
extern "C" void kernel_launch(void* const* d_in, const int* in_sizes, int n_in,
                              void* d_out, int out_size, void* d_ws, size_t ws_size,
                              hipStream_t stream) {
  const float* x    = (const float*)d_in[0];
  const float* te   = (const float*)d_in[1];
  const float* Wqkv = (const float*)d_in[2];
  const float* bqkv = (const float*)d_in[3];
  const float* Wt   = (const float*)d_in[4];
  const float* bt   = (const float*)d_in[5];
  const float* Wout = (const float*)d_in[6];
  const float* bout = (const float*)d_in[7];
  const float* Er   = (const float*)d_in[8];

  char* ws = (char*)d_ws;
  float* tqkv = (float*)ws;
  size_t off = (size_t)(128 << 10);
  const size_t QKV_BUF = (size_t)B_ * H_ * N_ * D_ * sizeof(half_t);   // 16 MB
  half_t* qh    = (half_t*)(ws + off); off += QKV_BUF;
  half_t* kh    = (half_t*)(ws + off); off += QKV_BUF;
  half_t* vTh   = (half_t*)(ws + off); off += QKV_BUF;
  half_t* oh    = (half_t*)(ws + off); off += QKV_BUF;
  half_t* xh    = (half_t*)(ws + off); off += (size_t)B_ * N_ * C_ * sizeof(half_t);
  half_t* Wqkvh = (half_t*)(ws + off); off += (size_t)C_ * 3 * C_ * sizeof(half_t);
  half_t* Wouth = (half_t*)(ws + off); off += (size_t)C_ * C_ * sizeof(half_t);
  half_t* Erh   = (half_t*)(ws + off); off += (size_t)N_ * D_ * sizeof(half_t);

  const int n_x = B_ * N_ * C_, n_wq = C_ * 3 * C_, n_wo = C_ * C_, n_er = N_ * D_;
  cvt_kernel<<<dim3((n_x  + 255) / 256), 256, 0, stream>>>(x,    xh,    n_x);
  cvt_kernel<<<dim3((n_wq + 255) / 256), 256, 0, stream>>>(Wqkv, Wqkvh, n_wq);
  cvt_kernel<<<dim3((n_wo + 255) / 256), 256, 0, stream>>>(Wout, Wouth, n_wo);
  cvt_kernel<<<dim3((n_er + 255) / 256), 256, 0, stream>>>(Er,   Erh,   n_er);

  time_proj_kernel<<<dim3((B_ * 3 * C_ + 255) / 256), 256, 0, stream>>>(te, Wt, bt, tqkv);
  qkv_gemm_kernel<<<dim3((B_ * N_) / 64, (3 * C_) / 64), 128, 0, stream>>>(
      xh, Wqkvh, bqkv, tqkv, qh, kh, vTh);
  attn_kernel<<<dim3(N_ / 16, H_, B_), 32, 0, stream>>>(qh, kh, vTh, Erh, oh);
  out_gemm_kernel<<<dim3((B_ * N_) / 64, C_ / 64), 128, 0, stream>>>(
      oh, Wouth, bout, (float*)d_out);
}
